// DynamicMoELayer_63608465653850
// MI455X (gfx1250) — matrix operand order repaired and checked by reading, not compiled, not run on hardware
//
#include <hip/hip_runtime.h>
#include <hip/hip_bf16.h>
#include <math.h>

#define T_TOK 8192
#define C_DIM 512
#define I_DIM 64
#define E_EXP 32

typedef __attribute__((ext_vector_type(16))) __bf16 v16bf;
typedef __attribute__((ext_vector_type(8)))  __bf16 v8bf;
typedef __attribute__((ext_vector_type(8)))  float  v8f;

// A-fragment (16x32 bf16, wave32): lane L holds row M=L%16, half=L/16.
// Elements i=0..7  come from k = half*16 + i          (contiguous 16B)
// Elements i=8..15 come from k = half*16 + i          (contiguous 16B)
__device__ __forceinline__ v16bf load_frag_a(const __bf16* p, int half) {
  v8bf lo = *(const v8bf*)(p + half * 16);
  v8bf hi = *(const v8bf*)(p + half * 16 + 8);
  return __builtin_shufflevector(lo, hi, 0,1,2,3,4,5,6,7,8,9,10,11,12,13,14,15);
}

// ---------------- Kernel 1: gating ----------------
// one wave per token, lane = expert
__global__ __launch_bounds__(256)
void moe_gate_kernel(const float* __restrict__ x, const float* __restrict__ sim,
                     const float* __restrict__ gates,
                     float* __restrict__ pre_out, float* __restrict__ mask_out,
                     float* __restrict__ rw_out) {
  __shared__ __align__(16) float Xl[8 * C_DIM];   // 16 KB
  __shared__ float colnorm[E_EXP];
  const int tid = threadIdx.x;
  const int t0  = blockIdx.x * 8;
  for (int idx = tid; idx < 8 * C_DIM; idx += 256) {
    int t = idx >> 9, c = idx & (C_DIM - 1);
    Xl[idx] = x[(size_t)(t0 + t) * C_DIM + c];
  }
  if (tid < E_EXP) {
    float s = 0.f;
    for (int c = 0; c < C_DIM; ++c) { float v = sim[c * E_EXP + tid]; s += v * v; }
    colnorm[tid] = sqrtf(s);
  }
  __syncthreads();
  const int wv = tid >> 5;           // token within block
  const int e  = tid & 31;           // expert lane
  const int trow = t0 + wv;
  float dot = 0.f;
  for (int c = 0; c < C_DIM; ++c)
    dot += Xl[wv * C_DIM + c] * sim[c * E_EXP + e];
  float xsq = 0.f;
  for (int c = e; c < C_DIM; c += 32) { float v = Xl[wv * C_DIM + c]; xsq += v * v; }
  for (int d = 16; d > 0; d >>= 1) xsq += __shfl_xor(xsq, d, 32);
  float xn = fmaxf(sqrtf(xsq), 1e-12f);
  float cn = fmaxf(colnorm[e], 1e-12f);
  float logit = dot / (xn * cn);
  float sg  = 1.f / (1.f + expf(-gates[e]));
  float pre = logit - sg;
  float gated = fmaxf(pre, 0.f);
  bool active = gated > 0.f;
  unsigned long long bal = __ballot(active);
  bool inactive = (bal == 0ull);
  int rank = 0;                       // top-k with tie-break by index
  for (int j = 0; j < 32; ++j) {
    float lj = __shfl(logit, j, 32);
    rank += (lj > logit) || (lj == logit && j < e);
  }
  float maskv = (inactive && rank < (E_EXP / 2)) ? 1.f : (active ? 1.f : 0.f);
  const float NEGV = -3.3895313892515355e38f;     // -max(bf16)
  float masked = (maskv > 0.f) ? gated : NEGV;
  float mx = masked;
  for (int d = 16; d > 0; d >>= 1) mx = fmaxf(mx, __shfl_xor(mx, d, 32));
  float ex = expf(masked - mx);
  float sm = ex;
  for (int d = 16; d > 0; d >>= 1) sm += __shfl_xor(sm, d, 32);
  float rw = ex / sm;
  pre_out [(size_t)trow * E_EXP + e] = pre;
  mask_out[(size_t)trow * E_EXP + e] = maskv;
  rw_out  [(size_t)trow * E_EXP + e] = rw;
}

// ---------------- Kernel 2: dense expert GEMMs (bf16 WMMA) ----------------
// Block: 256 threads (8 waves), 32-token tile staged ONCE, loops all 32 experts.
// GEMM1: H = gelu(X(32x512) @ W1[e](512x64)), W1 K-chunked by 128.
// GEMM2: Y = H(32x64) @ W2[e](64x512), W2 N-chunked by 128.
// feo/final written with non-temporal hints to keep X/W1/W2 L2-resident.
__global__ __launch_bounds__(256)
void moe_expert_kernel(const float* __restrict__ x,
                       const float* __restrict__ W1, const float* __restrict__ W2,
                       const float* __restrict__ mask_g, const float* __restrict__ rw_g,
                       float* __restrict__ final_g, float* __restrict__ feo_g) {
  __shared__ __align__(32) __bf16 Xs [32  * C_DIM]; // 32 KB  full X tile, [t][k]
  __shared__ __align__(32) __bf16 W1c[I_DIM * 128]; // 16 KB  W1 chunk, transposed [i][k]
  __shared__ __align__(32) __bf16 Hs [32  * I_DIM]; //  4 KB  H, [t][i]
  __shared__ __align__(32) __bf16 W2s[128 * I_DIM]; // 16 KB  W2 chunk, transposed [c'][i]
  const int tid  = threadIdx.x;
  const int t0   = blockIdx.x * 32;
  const int w    = tid >> 5, l = tid & 31;
  const int half = l >> 4, ln = l & 15;
  const int mt   = w >> 2;            // 0..1  (M tile)
  const int nt   = w & 3;             // 0..3  (N tile)

  // stage full X tile once (f32 -> bf16): reused by all 32 experts
  for (int idx = tid; idx < 32 * C_DIM; idx += 256) {
    int t = idx >> 9, c = idx & (C_DIM - 1);
    Xs[idx] = (__bf16)x[(size_t)(t0 + t) * C_DIM + c];
  }

  v8f facc[8];                         // final-output accumulators (4 chunks x 2 subtiles)
  for (int i = 0; i < 8; ++i)
    for (int r = 0; r < 8; ++r) facc[i][r] = 0.f;

  const int rbase = t0 + mt * 16 + half * 8;  // first global row this lane owns (r=0..7)

  for (int e = 0; e < E_EXP; ++e) {
    const float* W1e = W1 + (size_t)e * C_DIM * I_DIM;
    const float* W2e = W2 + (size_t)e * I_DIM * C_DIM;

    // prefetch next expert's weights into cache while this expert computes
    if (e + 1 < E_EXP) {
      const char* p1 = (const char*)(W1e + (size_t)C_DIM * I_DIM);
      const char* p2 = (const char*)(W2e + (size_t)I_DIM * C_DIM);
      __builtin_prefetch(p1 + tid * 512, 0, 1);
      __builtin_prefetch(p2 + tid * 512, 0, 1);
    }

    // ---- GEMM1: accumulate over K chunks of 128 (only W1 restaged) ----
    v8f acc = {};
    for (int kc = 0; kc < C_DIM; kc += 128) {
      __syncthreads();                                        // prev W1c consumers done
      for (int idx = tid; idx < I_DIM * 128; idx += 256) {    // W1 chunk, transposed
        int k = idx >> 6, i = idx & 63;
        W1c[i * 128 + k] = (__bf16)W1e[(size_t)(kc + k) * I_DIM + i];
      }
      __syncthreads();
      const __bf16* arow = &Xs [(mt * 16 + ln) * C_DIM + kc];
      const __bf16* bcol = &W1c[(nt * 16 + ln) * 128];
      for (int kk = 0; kk < 128; kk += 32) {
        v16bf a = load_frag_a(arow + kk, half);
        v16bf b = *(const v16bf*)(bcol + kk + half * 16);
        acc = __builtin_amdgcn_wmma_f32_16x16x32_bf16(false, a, false, b,
                                                      (short)0, acc, false, false);
      }
    }
    // gelu(exact) -> Hs (bf16)
    for (int r = 0; r < 8; ++r) {
      float hv = acc[r];
      hv = 0.5f * hv * (1.f + erff(hv * 0.70710678118654752f));
      Hs[(mt * 16 + half * 8 + r) * I_DIM + nt * 16 + ln] = (__bf16)hv;
    }
    // per-row mask & routing weight for this expert
    float maskv[8], rwv[8];
    for (int r = 0; r < 8; ++r) {
      maskv[r] = mask_g[(size_t)(rbase + r) * E_EXP + e];
      rwv[r]   = rw_g  [(size_t)(rbase + r) * E_EXP + e];
    }
    // ---- GEMM2 over N chunks of 128 ----
    for (int ch = 0; ch < 4; ++ch) {
      __syncthreads();                 // Hs visible (ch==0) / previous W2s consumed
      for (int idx = tid; idx < 128 * I_DIM; idx += 256) {
        int i = idx >> 7, cc = idx & 127;
        W2s[cc * I_DIM + i] = (__bf16)W2e[(size_t)i * C_DIM + ch * 128 + cc];
      }
      __syncthreads();
      for (int sub = 0; sub < 2; ++sub) {
        int ntile = nt + 4 * sub;
        v8f a2 = {};
        const __bf16* arow2 = &Hs [(mt * 16 + ln) * I_DIM];
        const __bf16* bcol2 = &W2s[(ntile * 16 + ln) * I_DIM];
        for (int kk = 0; kk < I_DIM; kk += 32) {
          v16bf a = load_frag_a(arow2 + kk, half);
          v16bf b = *(const v16bf*)(bcol2 + kk + half * 16);
          a2 = __builtin_amdgcn_wmma_f32_16x16x32_bf16(false, a, false, b,
                                                       (short)0, a2, false, false);
        }
        int col = ch * 128 + ntile * 16 + ln;
        for (int r = 0; r < 8; ++r) {
          float yv = a2[r] * maskv[r];
          // streaming 512MB output: non-temporal, don't pollute L2
          __builtin_nontemporal_store(
              yv, &feo_g[((size_t)(rbase + r) * E_EXP + e) * C_DIM + col]);
          facc[ch * 2 + sub][r] += rwv[r] * yv;
        }
      }
    }
  }
  // final_output: each (row, col) owned by exactly one lane -> plain NT stores
  for (int ch = 0; ch < 4; ++ch)
    for (int sub = 0; sub < 2; ++sub) {
      int ntile = nt + 4 * sub;
      int col = ch * 128 + ntile * 16 + ln;
      for (int r = 0; r < 8; ++r)
        __builtin_nontemporal_store(
            facc[ch * 2 + sub][r], &final_g[(size_t)(rbase + r) * C_DIM + col]);
    }
}

extern "C" void kernel_launch(void* const* d_in, const int* in_sizes, int n_in,
                              void* d_out, int out_size, void* d_ws, size_t ws_size,
                              hipStream_t stream) {
  (void)in_sizes; (void)n_in; (void)out_size; (void)ws_size;
  const float* x     = (const float*)d_in[0];  // (T, C)
  const float* sim   = (const float*)d_in[1];  // (C, E)
  const float* gates = (const float*)d_in[2];  // (E,)
  const float* W1    = (const float*)d_in[3];  // (E, C, I)
  const float* W2    = (const float*)d_in[4];  // (E, I, C)

  float* out      = (float*)d_out;
  float* final_g  = out;                                        // (T, C)
  float* feo_g    = final_g + (size_t)T_TOK * C_DIM;            // (T, E, C)
  float* pre_out  = feo_g   + (size_t)T_TOK * E_EXP * C_DIM;    // (T, E)
  float* mask_out = pre_out + (size_t)T_TOK * E_EXP;            // (T, E)
  float* rw_ws    = (float*)d_ws;                               // (T, E) scratch

  moe_gate_kernel<<<T_TOK / 8, 256, 0, stream>>>(x, sim, gates, pre_out, mask_out, rw_ws);
  moe_expert_kernel<<<T_TOK / 32, 256, 0, stream>>>(x, W1, W2, mask_out, rw_ws,
                                                    final_g, feo_g);
}
